// Model_72748156060319
// MI455X (gfx1250) — matrix operand order repaired and checked by reading, not compiled, not run on hardware
//
#include <hip/hip_runtime.h>
#include <hip/hip_bf16.h>

typedef __attribute__((ext_vector_type(16))) _Float16 v16h;
typedef __attribute__((ext_vector_type(8)))  float    v8f;

#define Ecnt   50000
#define Ntrip  1000000
#define Rrel   200
#define NRELc  401
#define Hdim   256
#define Thop   3
#define Lch    3
#define Bsz    8
#define TAUc   10.0f
#define TOPKc  1000
#define G4H    1024          // 4*H
#define MATSZ  262144        // halves per repacked 1024x256 matrix: 64*8*32*16

// ---------------------------------------------------------------------------
// Weight repack: f32 [l][4H][H] -> f16 WMMA B-fragment blobs.
// Blob layout per (l,dir,mat): [ntile(64)][ktile(8)][lane(32)][e(16)] halves.
// B-fragment (32x16 f16): lane<16 -> K=0..15, lane>=16 -> K=16..31,
// N = lane%16, consecutive halves = consecutive K.
// ---------------------------------------------------------------------------
__global__ void prep_weights(const float* __restrict__ Wih_f, const float* __restrict__ Whh_f,
                             const float* __restrict__ Wih_b, const float* __restrict__ Whh_b,
                             _Float16* __restrict__ blob) {
  long idx = (long)blockIdx.x * 256 + threadIdx.x;
  const long total = 12L * MATSZ;           // (l,dir,mat) = 3*2*2
  if (idx >= total) return;
  int e    = (int)(idx & 15);
  int lane = (int)((idx >> 4) & 31);
  int kt   = (int)((idx >> 9) & 7);
  int nt   = (int)((idx >> 12) & 63);
  int mf   = (int)(idx >> 18);              // ((l*2+dir)*2+mat)
  int mat  = mf & 1;
  int dir  = (mf >> 1) & 1;
  int l    = mf >> 2;
  const float* src = (dir == 0) ? (mat == 0 ? Wih_f : Whh_f)
                                : (mat == 0 ? Wih_b : Whh_b);
  int n = nt * 16 + (lane & 15);
  int k = kt * 32 + 16 * (lane >> 4) + e;
  blob[idx] = (_Float16)src[(long)l * G4H * Hdim + (long)n * Hdim + k];
}

// Build 16x32 f16 A-fragment from a [16][256] f16 LDS buffer (matches B blob):
// lane<16: K 0..15, lane>=16: K 16..31, consecutive halves = consecutive K.
__device__ __forceinline__ v16h build_a(const _Float16* buf, int ktile, int lane) {
  int row = lane & 15, hs = lane >> 4;
  v16h a;
#pragma unroll
  for (int e = 0; e < 16; ++e) {
    int j = e >> 1, hb = e & 1;
    int kl = 2 * j + hb;
    kl += 16 * hs;
    a[e] = buf[row * Hdim + ktile * 32 + kl];
  }
  return a;
}

__device__ __forceinline__ float sigm(float x) { return 1.0f / (1.0f + __expf(-x)); }

// ---------------------------------------------------------------------------
// One workgroup per (l, dir): WMMA input projection + 4 WMMA recurrent steps.
// A rows 0..7 = emb[input_r[b]], row 8 = emb[NREL-1] -> one pass gives both
// pre(q) and pre(emb_last). hseq layout: [l][dir][step][b][h] f32.
// ---------------------------------------------------------------------------
__global__ void __launch_bounds__(256)
lstm_kernel(const float* __restrict__ emb, const int* __restrict__ input_r,
            const float* __restrict__ bih_f, const float* __restrict__ bhh_f,
            const float* __restrict__ bih_b, const float* __restrict__ bhh_b,
            const _Float16* __restrict__ blob, float* __restrict__ hseq) {
  __shared__ _Float16 xg[8192];        // stage1: xbuf[16][256]; stage2: gbuf[8][1024]
  __shared__ _Float16 preq[8 * 1024];  // pre for q rows (f16)
  __shared__ _Float16 pree[1024];      // pre for emb[-1] row
  __shared__ _Float16 hbuf[16 * 256];  // h (rows 8..15 stay zero)
  __shared__ float    cbuf[8 * 256];

  const int tid = threadIdx.x, lane = tid & 31, wave = tid >> 5;
  const int bid = blockIdx.x, l = bid >> 1, dir = bid & 1;
  const float* bih = dir ? bih_b : bih_f;
  const float* bhh = dir ? bhh_b : bhh_f;
  const _Float16* wih = blob + (long)((l * 2 + dir) * 2 + 0) * MATSZ;
  const _Float16* whh = blob + (long)((l * 2 + dir) * 2 + 1) * MATSZ;

  for (int i = tid; i < 16 * 256; i += 256) {
    int row = i >> 8, k = i & 255;
    float v = (row < 8) ? emb[(long)input_r[row] * Hdim + k]
                        : emb[(long)(NRELc - 1) * Hdim + k];
    xg[i]  = (_Float16)v;
    hbuf[i] = (_Float16)0.0f;
  }
  for (int i = tid; i < 8 * 256; i += 256) cbuf[i] = 0.0f;
  __syncthreads();

  const int col = lane & 15, hs = lane >> 4;
  v16h areg[8];

  // ---- Stage 1: input projection, D -> preq/pree (+ bih + bhh) ----
#pragma unroll
  for (int kt = 0; kt < 8; ++kt) areg[kt] = build_a(xg, kt, lane);
  for (int j = 0; j < 8; ++j) {
    int nt = wave * 8 + j;
    v8f acc = {};
#pragma unroll
    for (int kt = 0; kt < 8; ++kt) {
      v16h bfrag = *(const v16h*)(wih + ((long)(nt * 8 + kt) * 32 + lane) * 16);
      acc = __builtin_amdgcn_wmma_f32_16x16x32_f16(false, areg[kt], false, bfrag,
                                                   (short)0, acc, false, false);
    }
    int n = nt * 16 + col;
    float bv = bih[l * G4H + n] + bhh[l * G4H + n];
#pragma unroll
    for (int r = 0; r < 8; ++r) {
      int M = r + 8 * hs;
      float v = acc[r] + bv;
      if (M < 8)       preq[M * 1024 + n] = (_Float16)v;
      else if (M == 8) pree[n]            = (_Float16)v;
    }
  }

  // ---- Stage 2: 4 recurrent steps ----
  for (int s = 0; s < 4; ++s) {
    __syncthreads();   // hbuf fresh; gbuf(xg) free to overwrite
#pragma unroll
    for (int kt = 0; kt < 8; ++kt) areg[kt] = build_a(hbuf, kt, lane);
    for (int j = 0; j < 8; ++j) {
      int nt = wave * 8 + j;
      v8f acc = {};
#pragma unroll
      for (int kt = 0; kt < 8; ++kt) {
        v16h bfrag = *(const v16h*)(whh + ((long)(nt * 8 + kt) * 32 + lane) * 16);
        acc = __builtin_amdgcn_wmma_f32_16x16x32_f16(false, areg[kt], false, bfrag,
                                                     (short)0, acc, false, false);
      }
      int n = nt * 16 + col;
#pragma unroll
      for (int r = 0; r < 8; ++r) {
        int M = r + 8 * hs;
        if (M < 8) xg[M * 1024 + n] = (_Float16)acc[r];
      }
    }
    __syncthreads();

    const bool useE = dir ? (s == 0) : (s == 3);  // bwd seq = [e,q,q,q]; fwd = [q,q,q,e]
    const int hcol = tid;
    for (int b = 0; b < 8; ++b) {
      float pi = useE ? (float)pree[hcol]       : (float)preq[b * 1024 + hcol];
      float pf = useE ? (float)pree[256 + hcol] : (float)preq[b * 1024 + 256 + hcol];
      float pg = useE ? (float)pree[512 + hcol] : (float)preq[b * 1024 + 512 + hcol];
      float po = useE ? (float)pree[768 + hcol] : (float)preq[b * 1024 + 768 + hcol];
      float gi = pi + (float)xg[b * 1024 + hcol];
      float gf = pf + (float)xg[b * 1024 + 256 + hcol];
      float gg = pg + (float)xg[b * 1024 + 512 + hcol];
      float go = po + (float)xg[b * 1024 + 768 + hcol];
      float cc = sigm(gf) * cbuf[b * 256 + hcol] + sigm(gi) * tanhf(gg);
      float hh = sigm(go) * tanhf(cc);
      cbuf[b * 256 + hcol] = cc;
      hbuf[b * 256 + hcol] = (_Float16)hh;
      hseq[((long)bid * 4 + s) * 2048 + b * 256 + hcol] = hh;
    }
  }
}

// w_all[b][t][l][n] = concat(hf[l][t], hb_rev[l][3-t]) . linW[n] + linb[n]
__global__ void linear_kernel(const float* __restrict__ hseq, const float* __restrict__ linW,
                              const float* __restrict__ linb, float* __restrict__ w_all) {
  int idx = blockIdx.x * 256 + threadIdx.x;
  if (idx >= Bsz * Thop * Lch * NRELc) return;
  int n = idx % NRELc;
  int l = (idx / NRELc) % Lch;
  int t = (idx / (NRELc * Lch)) % Thop;
  int b = idx / (NRELc * Lch * Thop);
  const float* hf = hseq + ((long)(l * 2 + 0) * 4 + t) * 2048 + b * 256;
  const float* hb = hseq + ((long)(l * 2 + 1) * 4 + (3 - t)) * 2048 + b * 256;
  const float* w0 = linW + (long)n * 512;
  float acc = linb[n];
  for (int k = 0; k < 256; ++k) acc += hf[k] * w0[k];
  for (int k = 0; k < 256; ++k) acc += hb[k] * w0[256 + k];
  w_all[idx] = acc;
}

__global__ void softmax_kernel(const float* __restrict__ w_all, float* __restrict__ wsoft) {
  int row = blockIdx.x * blockDim.x + threadIdx.x;
  if (row >= Bsz * Thop * Lch) return;
  const float* x = w_all + (long)row * NRELc;
  float* y = wsoft + (long)row * NRELc;
  float m = -1e30f;
  for (int n = 0; n < NRELc; ++n) m = fmaxf(m, x[n] / TAUc);
  float ssum = 0.0f;
  for (int n = 0; n < NRELc; ++n) { float e = __expf(x[n] / TAUc - m); y[n] = e; ssum += e; }
  float inv = 1.0f / ssum;
  for (int n = 0; n < NRELc; ++n) y[n] *= inv;
}

__global__ void init_x0(const int* __restrict__ input_x, float* __restrict__ SA) {
  long i = (long)blockIdx.x * 256 + threadIdx.x;
  if (i >= (long)Bsz * Lch * Ecnt) return;
  int e = (int)(i % Ecnt);
  int b = (int)(i / ((long)Lch * Ecnt));
  SA[i] = (e == input_x[b]) ? 1.0f : 0.0f;
}

// out = w_identity * xin  (also serves as the per-hop zero/base init)
__global__ void identity_init(const float* __restrict__ xin, const float* __restrict__ wsoft,
                              int t, float* __restrict__ out) {
  long i = (long)blockIdx.x * 256 + threadIdx.x;
  if (i >= (long)Bsz * Lch * Ecnt) return;
  int c = (int)(i / Ecnt);
  int b = c / Lch, l = c % Lch;
  float wid = wsoft[((long)(b * Thop + t) * Lch + l) * NRELc + (NRELc - 1)];
  out[i] = wid * xin[i];
}

// Scatter-gather over triples; skip atomics for the (>=98%) zero state entries.
__global__ void propagate_kernel(const int* __restrict__ heads, const int* __restrict__ tails,
                                 const int* __restrict__ tails2, const int* __restrict__ rels,
                                 const float* __restrict__ xin, const float* __restrict__ wsoft,
                                 int t, float* __restrict__ out) {
  int i = blockIdx.x * 256 + threadIdx.x;
  if (i >= Ntrip) return;
  int h = heads[i], tl = tails[i], t2 = tails2[i], r = rels[i];
#pragma unroll
  for (int b = 0; b < Bsz; ++b) {
    const float* wrow = wsoft + (long)(b * Thop + t) * Lch * NRELc;
#pragma unroll
    for (int l = 0; l < Lch; ++l) {
      long base = (long)(b * Lch + l) * Ecnt;
      float xf = xin[base + h];
      float xb = xin[base + t2];
      if (xf != 0.0f) atomicAdd(out + base + tl, xf * wrow[l * NRELc + r]);
      if (xb != 0.0f) atomicAdd(out + base + h,  xb * wrow[l * NRELc + r + Rrel]);
    }
  }
}

__global__ void normalize_kernel(float* __restrict__ s) {
  __shared__ float red[256];
  float* p = s + (long)blockIdx.x * Ecnt;
  float acc = 0.0f;
  for (int e = threadIdx.x; e < Ecnt; e += 256) acc += p[e];
  red[threadIdx.x] = acc;
  __syncthreads();
  for (int st = 128; st > 0; st >>= 1) {
    if (threadIdx.x < st) red[threadIdx.x] += red[threadIdx.x + st];
    __syncthreads();
  }
  float inv = 1.0f / fmaxf(red[0], 1e-7f);
  for (int e = threadIdx.x; e < Ecnt; e += 256) p[e] *= inv;
}

__device__ __forceinline__ unsigned fkey(float x) {
  unsigned u = __float_as_uint(x);
  return (u & 0x80000000u) ? ~u : (u | 0x80000000u);
}

// Exact k-th largest via 4-pass radix select on order-preserving keys,
// then threshold with x >= th (reference tie semantics).
__global__ void topk_kernel(const float* __restrict__ src, float* __restrict__ dst) {
  __shared__ unsigned hist[256];
  __shared__ unsigned sprefix;
  __shared__ int sremain;
  const float* p = src + (long)blockIdx.x * Ecnt;
  float* q = dst + (long)blockIdx.x * Ecnt;
  if (threadIdx.x == 0) { sprefix = 0u; sremain = TOPKc; }
  __syncthreads();
  for (int pass = 3; pass >= 0; --pass) {
    hist[threadIdx.x] = 0u;
    __syncthreads();
    int shift = pass * 8;
    unsigned pref = sprefix;
    for (int e = threadIdx.x; e < Ecnt; e += 256) {
      unsigned k = fkey(p[e]);
      bool match = (pass == 3) || ((k >> (shift + 8)) == (pref >> (shift + 8)));
      if (match) atomicAdd(&hist[(k >> shift) & 255u], 1u);
    }
    __syncthreads();
    if (threadIdx.x == 0) {
      int rem = sremain;
      unsigned cum = 0;
      int d = 255;
      for (; d >= 0; --d) { cum += hist[d]; if ((int)cum >= rem) break; }
      if (d < 0) d = 0;
      sremain = rem - (int)(cum - hist[d]);
      sprefix = pref | ((unsigned)d << shift);
    }
    __syncthreads();
  }
  unsigned th = sprefix;
  for (int e = threadIdx.x; e < Ecnt; e += 256) {
    float v = p[e];
    q[e] = (fkey(v) >= th) ? v : 0.0f;
  }
}

__global__ void final_sum(const float* __restrict__ s, float* __restrict__ out) {
  long i = (long)blockIdx.x * 256 + threadIdx.x;
  if (i >= (long)Bsz * Ecnt) return;
  int e = (int)(i % Ecnt);
  int b = (int)(i / Ecnt);
  const float* p = s + (long)b * Lch * Ecnt + e;
  out[i] = p[0] + p[(long)Ecnt] + p[2L * Ecnt];
}

extern "C" void kernel_launch(void* const* d_in, const int* in_sizes, int n_in,
                              void* d_out, int out_size, void* d_ws, size_t ws_size,
                              hipStream_t stream) {
  (void)in_sizes; (void)n_in; (void)out_size; (void)ws_size;
  const int*   input_x = (const int*)d_in[0];
  const int*   input_r = (const int*)d_in[1];
  const int*   e2t     = (const int*)d_in[2];   // [3][N]
  const int*   t2e     = (const int*)d_in[3];   // [2][N]
  const int*   r2t     = (const int*)d_in[4];   // [1][N]
  const float* emb     = (const float*)d_in[5];
  const float* Wih_f   = (const float*)d_in[6];
  const float* Whh_f   = (const float*)d_in[7];
  const float* bih_f   = (const float*)d_in[8];
  const float* bhh_f   = (const float*)d_in[9];
  const float* Wih_b   = (const float*)d_in[10];
  const float* Whh_b   = (const float*)d_in[11];
  const float* bih_b   = (const float*)d_in[12];
  const float* bhh_b   = (const float*)d_in[13];
  const float* linW    = (const float*)d_in[14];
  const float* linb    = (const float*)d_in[15];
  float* out = (float*)d_out;

  const int* heads  = e2t;                // e2triple[0]
  const int* tails2 = e2t + 2L * Ntrip;   // e2triple[2]
  const int* tails  = t2e + 1L * Ntrip;   // triple2e[1]
  const int* rels   = r2t;                // r2triple[0]

  // Workspace carve-up (256B aligned)
  char* ws = (char*)d_ws;
  size_t cur = 0;
  auto alloc = [&](size_t bytes) { size_t o = (cur + 255) & ~(size_t)255; cur = o + bytes; return o; };
  _Float16* blob  = (_Float16*)(ws + alloc(12L * MATSZ * sizeof(_Float16)));          // 6 MB
  float*    hseq  = (float*)   (ws + alloc(6L * 4 * 8 * 256 * sizeof(float)));        // 192 KB
  float*    w_all = (float*)   (ws + alloc((size_t)Bsz * Thop * Lch * NRELc * 4));
  float*    wsoft = (float*)   (ws + alloc((size_t)Bsz * Thop * Lch * NRELc * 4));
  float*    SA    = (float*)   (ws + alloc((size_t)Bsz * Lch * Ecnt * 4));            // 4.8 MB
  float*    SB    = (float*)   (ws + alloc((size_t)Bsz * Lch * Ecnt * 4));            // 4.8 MB

  const long BLE = (long)Bsz * Lch * Ecnt;
  const int nBLE = (int)((BLE + 255) / 256);

  // 1. Dense matrix phase (WMMA)
  prep_weights<<<(int)((12L * MATSZ + 255) / 256), 256, 0, stream>>>(Wih_f, Whh_f, Wih_b, Whh_b, blob);
  lstm_kernel<<<6, 256, 0, stream>>>(emb, input_r, bih_f, bhh_f, bih_b, bhh_b, blob, hseq);
  linear_kernel<<<(Bsz * Thop * Lch * NRELc + 255) / 256, 256, 0, stream>>>(hseq, linW, linb, w_all);
  softmax_kernel<<<1, 128, 0, stream>>>(w_all, wsoft);

  // 2. Sparse propagation phase (L2-resident gathers, guarded atomics)
  init_x0<<<nBLE, 256, 0, stream>>>(input_x, SA);
  for (int t = 0; t < Thop; ++t) {
    if (t > 0) topk_kernel<<<Bsz * Lch, 256, 0, stream>>>(SB, SA);
    identity_init<<<nBLE, 256, 0, stream>>>(SA, wsoft, t, SB);
    propagate_kernel<<<(Ntrip + 255) / 256, 256, 0, stream>>>(heads, tails, tails2, rels,
                                                              SA, wsoft, t, SB);
    normalize_kernel<<<Bsz * Lch, 256, 0, stream>>>(SB);
  }
  final_sum<<<(int)(((long)Bsz * Ecnt + 255) / 256), 256, 0, stream>>>(SB, out);
}
